// Net_37847251812678
// MI455X (gfx1250) — matrix operand order repaired and checked by reading, not compiled, not run on hardware
//
#include <hip/hip_runtime.h>

#define N_NODES 50000
#define IN_C    500
#define HID     128
#define OUT_C   64
#define E_EDGES 200000
#define E2      (E_EDGES + N_NODES)

typedef __attribute__((ext_vector_type(16))) __bf16 v16bf;
typedef __attribute__((ext_vector_type(8)))  __bf16 v8bf;
typedef __attribute__((ext_vector_type(4)))  __bf16 v4bf;
typedef __attribute__((ext_vector_type(8)))  float  v8f;
typedef __attribute__((ext_vector_type(4)))  unsigned int u32x4;
typedef __attribute__((ext_vector_type(8)))  int i32x8;
typedef __attribute__((ext_vector_type(4)))  int i32x4;

#if defined(__gfx1250__) && __has_builtin(__builtin_amdgcn_tensor_load_to_lds)
#define USE_TDM 1
#else
#define USE_TDM 0
#endif

static __device__ __forceinline__ v8f zero8() {
  v8f z = {0.f, 0.f, 0.f, 0.f, 0.f, 0.f, 0.f, 0.f};
  return z;
}

// Intra-wave LDS visibility (block == 1 wave): s_wait_dscnt 0 instead of a
// full barrier. asm memory clobbers stop the compiler reordering LDS ops.
static __device__ __forceinline__ void wave_lds_fence() {
  __asm__ volatile("" ::: "memory");
#if defined(__gfx1250__) && __has_builtin(__builtin_amdgcn_s_wait_dscnt)
  __builtin_amdgcn_s_wait_dscnt(0);
#else
  __syncthreads();
#endif
  __asm__ volatile("" ::: "memory");
}

// A fragment (16x32 bf16) from bf16 memory; per-lane = two 16B runs.
static __device__ __forceinline__ v16bf frag_a(const __bf16* p, int ld, int lane) {
  int row = lane & 15;
  int kb  = (lane >> 4) << 3;
  v8bf lo = *(const v8bf*)(p + row * ld + kb);
  v8bf hi = *(const v8bf*)(p + row * ld + 16 + kb);
  return __builtin_shufflevector(lo, hi, 0, 1, 2, 3, 4, 5, 6, 7,
                                 8, 9, 10, 11, 12, 13, 14, 15);
}

// A fragment with real K=16: one b128 + zero upper half.
static __device__ __forceinline__ v16bf frag_a_k16(const __bf16* p, int ld, int lane) {
  int row = lane & 15;
  int kb  = (lane >> 4) << 3;
  v8bf lo = *(const v8bf*)(p + row * ld + kb);
  v8bf zz = {};
  return __builtin_shufflevector(lo, zz, 0, 1, 2, 3, 4, 5, 6, 7,
                                 8, 9, 10, 11, 12, 13, 14, 15);
}

// Pre-swizzled 16x16 weight fragment (all 32 lanes stored; elements 8..15 zero).
static __device__ __forceinline__ v16bf wfrag16(const __bf16* wp, int g, int lane) {
  return *(const v16bf*)(wp + ((g * 32 + lane) << 4));
}

#if USE_TDM
// TDM gather: 16 rows of 256B (128 bf16), 16-bit row indices from LDS, into
// LDS at lds_off with 16B pad after each row (272B stride, bank-derotating).
static __device__ __forceinline__ void tdm_gather16(unsigned lds_off,
                                                    const __bf16* gbase,
                                                    const int* idx_lds) {
  unsigned long long ga = (unsigned long long)(size_t)gbase;
  u32x4 g0;
  g0[0] = 0x80000001u;                       // count=1, gather_mode=1, idx16
  g0[1] = lds_off;                           // lds_addr
  g0[2] = (unsigned)(ga & 0xFFFFFFFFu);      // global_addr lo
  g0[3] = (unsigned)((ga >> 32) & 0x01FFFFFFu) | (2u << 30);  // addr hi | type=2
  i32x8 g1;
  g1[0] = (3 << 16) | (1 << 20) | (5 << 22) | (3 << 25);  // 8B elems, pad 4dw/64dw
  g1[1] = (int)(32u << 16);                  // tensor_dim0 = 32 units (256B row)
  g1[2] = (int)((N_NODES & 0xFFFF) << 16);   // tensor_dim1 lo16
  g1[3] = (int)((N_NODES >> 16) & 0xFFFF) | (int)(32u << 16);  // dim1 hi | tile_dim0
  g1[4] = 16;                                // tile_dim1 = #indices
  g1[5] = 32;                                // tensor_dim0_stride = 32 units
  g1[6] = 0;
  g1[7] = 0;
  i32x4 g2, g3;
#pragma unroll
  for (int i = 0; i < 4; ++i) {
    g2[i] = (idx_lds[2 * i] & 0xFFFF) | (idx_lds[2 * i + 1] << 16);
    g3[i] = (idx_lds[8 + 2 * i] & 0xFFFF) | (idx_lds[8 + 2 * i + 1] << 16);
  }
#if __clang_major__ >= 23
  i32x8 g4 = {};
  __builtin_amdgcn_tensor_load_to_lds(g0, g1, g2, g3, g4, 0);
#else
  __builtin_amdgcn_tensor_load_to_lds(g0, g1, g2, g3, 0);
#endif
}
#endif

// ---------------- degree / normalization ----------------
__global__ void k_deg_init(float* deg, int n) {
  int i = blockIdx.x * blockDim.x + threadIdx.x;
  if (i < n) deg[i] = 1.0f;  // self loop
}
__global__ void k_deg_accum(float* deg, const int* colp, int e) {
  int i = blockIdx.x * blockDim.x + threadIdx.x;
  if (i < e) atomicAdd(&deg[colp[i]], 1.0f);
}
__global__ void k_dinv(float* deg, int n) {
  int i = blockIdx.x * blockDim.x + threadIdx.x;
  if (i < n) {
    float d = deg[i];
    deg[i] = d > 0.f ? rsqrtf(d) : 0.f;
  }
}
__global__ void k_zero4(float* p, int n4) {
  int i = blockIdx.x * blockDim.x + threadIdx.x;
  if (i < n4) ((float4*)p)[i] = make_float4(0.f, 0.f, 0.f, 0.f);
}

// ---------------- weight prep: f32 -> bf16 fragment layout ----------------
// qkv: [mat(3)][layer(3)][g(8)][lane(32)][t(16)]; t>=8 are zeros.
__global__ void k_prep_qkv(const float* __restrict__ Wq, const float* __restrict__ Wk,
                           const float* __restrict__ Wv, __bf16* __restrict__ dst) {
  int idx = blockIdx.x * blockDim.x + threadIdx.x;
  if (idx >= 3 * 3 * 8 * 512) return;
  int t = idx & 15;
  int ln = (idx >> 4) & 31;
  int g = (idx >> 9) & 7;
  int l = (idx >> 12) % 3;
  int mat = idx / 12288;
  const float* W = (mat == 0) ? Wq : (mat == 1) ? Wk : Wv;
  float v = (t < 8) ? W[(l * 8 + g) * 256 + (((ln >> 4) << 3) + t) * 16 + (ln & 15)] : 0.f;
  dst[idx] = (__bf16)v;
}

// W1 fragments: [kc(16)][ct(8)][lane(32)][t(16)], K padded 500->512 with zeros.
__global__ void k_prep_w1(const float* __restrict__ W1, __bf16* __restrict__ dst) {
  int idx = blockIdx.x * blockDim.x + threadIdx.x;
  if (idx >= 65536) return;
  int t = idx & 15;
  int lane = (idx >> 4) & 31;
  int ct = (idx >> 9) & 7;
  int kc = idx >> 12;
  int k = kc * 32 + ((t >> 3) << 4) + ((lane >> 4) << 3) + (t & 7);
  int c = ct * 16 + (lane & 15);
  float v = (k < IN_C) ? W1[(long)k * HID + c] : 0.f;
  dst[idx] = (__bf16)v;
}

// W2 fragments: [kc(4)][ct(4)][lane(32)][t(16)]
__global__ void k_prep_w2(const float* __restrict__ W2, __bf16* __restrict__ dst) {
  int idx = blockIdx.x * blockDim.x + threadIdx.x;
  if (idx >= 8192) return;
  int t = idx & 15;
  int lane = (idx >> 4) & 31;
  int ct = (idx >> 9) & 3;
  int kc = idx >> 11;
  int k = kc * 32 + ((t >> 3) << 4) + ((lane >> 4) << 3) + (t & 7);
  int c = ct * 16 + (lane & 15);
  dst[idx] = (__bf16)W2[k * OUT_C + c];
}

// ---------------- GEMM1: h = relu(X @ W1 + b1) -> bf16 slot0 ----------------
// LDS-free: A fragments straight from global f32 (rows private to each wave),
// B fragments from pre-swizzled w1p. block = 128 threads, 64 rows.
__global__ void k_gemm1(const float* __restrict__ X, const __bf16* __restrict__ w1p,
                        const float* __restrict__ b1, __bf16* __restrict__ h) {
  const int tid  = threadIdx.x;
  const int lane = tid & 31;
  const int wave = tid >> 5;
  const int row0 = blockIdx.x * 64;
  const int kb   = (lane >> 4) << 3;

  int gr = row0 + wave * 16 + (lane & 15);
  if (gr >= N_NODES) gr = N_NODES - 1;  // clamp: invalid rows only feed invalid outputs
  const float* rp = X + (long)gr * IN_C;

  v8f acc[8];
#pragma unroll
  for (int t = 0; t < 8; ++t) acc[t] = zero8();

  for (int kc = 0; kc < 15; ++kc) {  // full K chunks (0..479)
    int kk = kc * 32;
    const float4* q0 = (const float4*)(rp + kk + kb);
    const float4* q1 = (const float4*)(rp + kk + 16 + kb);
    float4 x0 = q0[0], x1 = q0[1], y0 = q1[0], y1 = q1[1];
    v16bf a;
    a[0] = (__bf16)x0.x;  a[1] = (__bf16)x0.y;  a[2] = (__bf16)x0.z;  a[3] = (__bf16)x0.w;
    a[4] = (__bf16)x1.x;  a[5] = (__bf16)x1.y;  a[6] = (__bf16)x1.z;  a[7] = (__bf16)x1.w;
    a[8] = (__bf16)y0.x;  a[9] = (__bf16)y0.y;  a[10] = (__bf16)y0.z; a[11] = (__bf16)y0.w;
    a[12] = (__bf16)y1.x; a[13] = (__bf16)y1.y; a[14] = (__bf16)y1.z; a[15] = (__bf16)y1.w;
#pragma unroll
    for (int ct = 0; ct < 8; ++ct) {
      v16bf b = *(const v16bf*)(w1p + (((kc * 8 + ct) * 32 + lane) << 4));
      acc[ct] = __builtin_amdgcn_wmma_f32_16x16x32_bf16(false, a, false, b,
                                                        (short)0, acc[ct], false, false);
    }
  }
  {  // K tail chunk kc=15 (480..499 valid of 480..511); branchless clamp+select
    const float4* q0 = (const float4*)(rp + 480 + kb);
    float4 x0 = q0[0], x1 = q0[1];
    v16bf a;
    a[0] = (__bf16)x0.x; a[1] = (__bf16)x0.y; a[2] = (__bf16)x0.z; a[3] = (__bf16)x0.w;
    a[4] = (__bf16)x1.x; a[5] = (__bf16)x1.y; a[6] = (__bf16)x1.z; a[7] = (__bf16)x1.w;
#pragma unroll
    for (int t = 0; t < 8; ++t) {
      int K = 496 + kb + t;
      int Kc = K < IN_C ? K : IN_C - 1;
      float v = rp[Kc];
      if (K >= IN_C) v = 0.f;
      a[8 + t] = (__bf16)v;
    }
#pragma unroll
    for (int ct = 0; ct < 8; ++ct) {
      v16bf b = *(const v16bf*)(w1p + (((15 * 8 + ct) * 32 + lane) << 4));
      acc[ct] = __builtin_amdgcn_wmma_f32_16x16x32_bf16(false, a, false, b,
                                                        (short)0, acc[ct], false, false);
    }
  }

  const int half = lane >> 4, coln = lane & 15;
  if (row0 + 64 <= N_NODES) {  // uniform fast path: no store guards
#pragma unroll
    for (int ct = 0; ct < 8; ++ct) {
      int gc = ct * 16 + coln;
      float bias = b1[gc];
#pragma unroll
      for (int r = 0; r < 8; ++r) {
        int grow = row0 + wave * 16 + r + 8 * half;
        float v = acc[ct][r] + bias;
        h[(long)grow * HID + gc] = (__bf16)fmaxf(v, 0.f);
      }
    }
  } else {
#pragma unroll
    for (int ct = 0; ct < 8; ++ct) {
      int gc = ct * 16 + coln;
      float bias = b1[gc];
#pragma unroll
      for (int r = 0; r < 8; ++r) {
        int grow = row0 + wave * 16 + r + 8 * half;
        if (grow < N_NODES) {
          float v = acc[ct][r] + bias;
          h[(long)grow * HID + gc] = (__bf16)fmaxf(v, 0.f);
        }
      }
    }
  }
}

// ---------------- attention layer: one wave per 16-edge tile ----------------
__global__ void k_attn(const __bf16* __restrict__ xall, float* __restrict__ agg,
                       const float* __restrict__ dinv,
                       const int* __restrict__ rowp, const int* __restrict__ colp,
                       const __bf16* __restrict__ wqp, const __bf16* __restrict__ wkp,
                       const __bf16* __restrict__ wvp,
                       const float* __restrict__ bq, const float* __restrict__ bk,
                       const float* __restrict__ bv, int l) {
  const int L = l + 1;
  __shared__ alignas(16) __bf16 qtile[16][136];        // 272B rows
  __shared__ alignas(16) __bf16 kvtile[3][16][136];
  __shared__ float scores[16][4];
  __shared__ float ew_s[16];
  __shared__ int   rnode_s[16], cnode_s[16];

  const int  lane = threadIdx.x;
  const long e0   = (long)blockIdx.x * 16;

  {  // both half-waves compute/write identical values: no divergence
    int li = lane & 15;
    long e = e0 + li;  // grid is exact: e < E2 always
    int rn = (e < E_EDGES) ? rowp[e] : (int)(e - E_EDGES);
    int cn = (e < E_EDGES) ? colp[e] : (int)(e - E_EDGES);
    rnode_s[li] = rn;
    cnode_s[li] = cn;
    ew_s[li] = dinv[rn] * dinv[cn];
  }
  wave_lds_fence();

#if USE_TDM
  tdm_gather16((unsigned)(size_t)&qtile[0][0],
               xall + (long)l * N_NODES * HID, cnode_s);
  for (int s = 0; s < L; ++s) {
    tdm_gather16((unsigned)(size_t)&kvtile[s][0][0],
                 xall + (long)s * N_NODES * HID, rnode_s);
  }
  __asm__ volatile("" ::: "memory");
  __builtin_amdgcn_s_wait_tensorcnt(0);
  __asm__ volatile("" ::: "memory");
#else
#pragma unroll
  for (int it = 0; it < 8; ++it) {
    int idx = lane + it * 32;
    int ed = idx >> 4, v = idx & 15;
    const uint4* src = (const uint4*)(xall + ((long)l * N_NODES + cnode_s[ed]) * HID);
    *(uint4*)&qtile[ed][v * 8] = src[v];
  }
  for (int s = 0; s < L; ++s) {
#pragma unroll
    for (int it = 0; it < 8; ++it) {
      int idx = lane + it * 32;
      int ed = idx >> 4, v = idx & 15;
      const uint4* src = (const uint4*)(xall + ((long)s * N_NODES + rnode_s[ed]) * HID);
      *(uint4*)&kvtile[s][ed][v * 8] = src[v];
    }
  }
  wave_lds_fence();
#endif

  const int half = lane >> 4, coln = lane & 15;
  for (int g = 0; g < 8; ++g) {
    // q = qin_g @ Wq[g] + bq
    v16bf aq  = frag_a_k16(&qtile[0][g * 16], 136, lane);
    v16bf wqf = wfrag16(wqp, g, lane);
    v8f qd = zero8();
    qd = __builtin_amdgcn_wmma_f32_16x16x32_bf16(false, aq, false, wqf,
                                                 (short)0, qd, false, false);
    float qbias = bq[g * 16 + coln];
#pragma unroll
    for (int r = 0; r < 8; ++r) qd[r] += qbias;

    // scores over slots
    v16bf wkf = wfrag16(wkp, g, lane);
    float kbias = bk[g * 16 + coln];
    for (int s = 0; s < L; ++s) {
      v16bf ak = frag_a_k16(&kvtile[s][0][g * 16], 136, lane);
      v8f kd = zero8();
      kd = __builtin_amdgcn_wmma_f32_16x16x32_bf16(false, ak, false, wkf,
                                                   (short)0, kd, false, false);
      float pr[8];
#pragma unroll
      for (int r = 0; r < 8; ++r) pr[r] = qd[r] * (kd[r] + kbias);
#pragma unroll
      for (int m = 1; m < 16; m <<= 1) {
#pragma unroll
        for (int r = 0; r < 8; ++r) pr[r] += __shfl_xor(pr[r], m, 32);
      }
      if (coln == 0) {
#pragma unroll
        for (int r = 0; r < 8; ++r) scores[r + 8 * half][s] = pr[r] * 0.25f;
      }
    }
    wave_lds_fence();

    // restricted softmax, duplicated across half-waves (same values written)
    {
      int row = lane & 15;
      float mx = 0.f;
      for (int s = 0; s < L; ++s) mx = fmaxf(mx, scores[row][s]);
      float denom = __expf(-mx);
      float es[3];
      for (int s = 0; s < L; ++s) { es[s] = __expf(scores[row][s] - mx); denom += es[s]; }
      float inv = 1.f / denom;
      for (int s = 0; s < L; ++s) scores[row][s] = es[s] * inv;
    }
    wave_lds_fence();

    // out = sum_s att * (kvin_s @ Wv[g] + bv)
    float og[8];
#pragma unroll
    for (int r = 0; r < 8; ++r) og[r] = 0.f;
    v16bf wvf = wfrag16(wvp, g, lane);
    float vbias = bv[g * 16 + coln];
    for (int s = 0; s < L; ++s) {
      v16bf av = frag_a_k16(&kvtile[s][0][g * 16], 136, lane);
      v8f vd = zero8();
      vd = __builtin_amdgcn_wmma_f32_16x16x32_bf16(false, av, false, wvf,
                                                   (short)0, vd, false, false);
#pragma unroll
      for (int r = 0; r < 8; ++r) {
        int m = r + 8 * half;
        og[r] += scores[m][s] * (vd[r] + vbias);
      }
    }

    // scatter: agg[col[e], g*16+coln] += ew * out
    int gch = g * 16 + coln;
#pragma unroll
    for (int r = 0; r < 8; ++r) {
      int m = r + 8 * half;
      atomicAdd(&agg[(long)cnode_s[m] * HID + gch], ew_s[m] * og[r]);
    }
  }
}

// relu(agg) -> bf16 slot (vectorized x4), and re-zero agg for next layer
__global__ void k_relu_reset(float* __restrict__ agg, __bf16* __restrict__ dst, int n4) {
  int i = blockIdx.x * blockDim.x + threadIdx.x;
  if (i < n4) {
    float4 v = ((float4*)agg)[i];
    v4bf o;
    o[0] = (__bf16)fmaxf(v.x, 0.f);
    o[1] = (__bf16)fmaxf(v.y, 0.f);
    o[2] = (__bf16)fmaxf(v.z, 0.f);
    o[3] = (__bf16)fmaxf(v.w, 0.f);
    *(v4bf*)(dst + i * 4) = o;
    ((float4*)agg)[i] = make_float4(0.f, 0.f, 0.f, 0.f);
  }
}

// ---------------- GEMM2 + log_softmax: one wave per 16 rows ----------------
__global__ void k_gemm2(const __bf16* __restrict__ hlast, const __bf16* __restrict__ w2p,
                        const float* __restrict__ b2, float* __restrict__ out) {
  __shared__ float logits[16][68];
  const int lane = threadIdx.x;
  const int row0 = blockIdx.x * 16;
  const int half = lane >> 4, coln = lane & 15;

  v8f acc[4];
#pragma unroll
  for (int t = 0; t < 4; ++t) acc[t] = zero8();

  const __bf16* Ap = hlast + (long)row0 * HID;
#pragma unroll
  for (int kc = 0; kc < 4; ++kc) {
    v16bf a = frag_a(Ap + kc * 32, HID, lane);
#pragma unroll
    for (int ct = 0; ct < 4; ++ct) {
      v16bf b = *(const v16bf*)(w2p + ((((kc * 4 + ct) * 32) + lane) << 4));
      acc[ct] = __builtin_amdgcn_wmma_f32_16x16x32_bf16(false, a, false, b,
                                                        (short)0, acc[ct], false, false);
    }
  }
#pragma unroll
  for (int ct = 0; ct < 4; ++ct) {
    int c = ct * 16 + coln;
    float bias = b2[c];
#pragma unroll
    for (int r = 0; r < 8; ++r) logits[r + 8 * half][c] = acc[ct][r] + bias;
  }
  wave_lds_fence();
  {  // duplicated across half-waves; duplicate identical stores are benign
    int row = lane & 15;
    int grow = row0 + row;
    float mx = -1e30f;
    for (int c = 0; c < OUT_C; ++c) mx = fmaxf(mx, logits[row][c]);
    float s = 0.f;
    for (int c = 0; c < OUT_C; ++c) s += __expf(logits[row][c] - mx);
    float lse = mx + __logf(s);
    for (int c = 0; c < OUT_C; ++c) out[(long)grow * OUT_C + c] = logits[row][c] - lse;
  }
}

extern "C" void kernel_launch(void* const* d_in, const int* in_sizes, int n_in,
                              void* d_out, int out_size, void* d_ws, size_t ws_size,
                              hipStream_t stream) {
  (void)in_sizes; (void)n_in; (void)out_size; (void)ws_size;
  const float* x  = (const float*)d_in[0];
  const int*   ei = (const int*)d_in[1];
  const float* W1 = (const float*)d_in[2];
  const float* b1 = (const float*)d_in[3];
  const float* Wq = (const float*)d_in[4];
  const float* Wk = (const float*)d_in[5];
  const float* Wv = (const float*)d_in[6];
  const float* bq = (const float*)d_in[7];
  const float* bk = (const float*)d_in[8];
  const float* bv = (const float*)d_in[9];
  const float* W2 = (const float*)d_in[10];
  const float* b2 = (const float*)d_in[11];
  float* out = (float*)d_out;

  char* ws = (char*)d_ws;
  __bf16* xall = (__bf16*)ws;                                   // 4 slots, bf16
  size_t xall_bytes = (size_t)4 * N_NODES * HID * sizeof(__bf16);
  float* agg = (float*)(ws + xall_bytes);                       // N x 128 f32
  size_t agg_bytes = (size_t)N_NODES * HID * sizeof(float);
  float* dinv = (float*)(ws + xall_bytes + agg_bytes);          // N f32
  size_t dinv_bytes = (size_t)N_NODES * sizeof(float);
  __bf16* wqkvp = (__bf16*)(ws + xall_bytes + agg_bytes + dinv_bytes);  // 36864
  __bf16* w1p = wqkvp + 36864;                                  // 65536
  __bf16* w2p = w1p + 65536;                                    // 8192

  const int* rowp = ei;
  const int* colp = ei + E_EDGES;

  k_deg_init<<<(N_NODES + 255) / 256, 256, 0, stream>>>(dinv, N_NODES);
  k_deg_accum<<<(E_EDGES + 255) / 256, 256, 0, stream>>>(dinv, colp, E_EDGES);
  k_dinv<<<(N_NODES + 255) / 256, 256, 0, stream>>>(dinv, N_NODES);
  k_zero4<<<(N_NODES * HID / 4 + 255) / 256, 256, 0, stream>>>(agg, N_NODES * HID / 4);
  k_prep_qkv<<<(36864 + 255) / 256, 256, 0, stream>>>(Wq, Wk, Wv, wqkvp);
  k_prep_w1<<<(65536 + 255) / 256, 256, 0, stream>>>(W1, w1p);
  k_prep_w2<<<(8192 + 255) / 256, 256, 0, stream>>>(W2, w2p);
  k_gemm1<<<(N_NODES + 63) / 64, 128, 0, stream>>>(x, w1p, b1, xall);

  for (int l = 0; l < 3; ++l) {
    k_attn<<<E2 / 16, 32, 0, stream>>>(
        xall, agg, dinv, rowp, colp,
        wqkvp + 0 * 12288 + l * 4096,
        wqkvp + 1 * 12288 + l * 4096,
        wqkvp + 2 * 12288 + l * 4096,
        bq + (size_t)l * HID, bk + (size_t)l * HID, bv + (size_t)l * HID, l);
    k_relu_reset<<<(N_NODES * HID / 4 + 255) / 256, 256, 0, stream>>>(
        agg, xall + (size_t)(l + 1) * N_NODES * HID, N_NODES * HID / 4);
  }
  k_gemm2<<<N_NODES / 16, 32, 0, stream>>>(xall + (size_t)3 * N_NODES * HID, w2p, b2, out);
}
